// source_encoder_convlstm_61907658605201
// MI455X (gfx1250) — matrix operand order repaired and checked by reading, not compiled, not run on hardware
//
#include <hip/hip_runtime.h>
#include <hip/hip_bf16.h>
#include <hip/hip_fp16.h>

typedef __attribute__((ext_vector_type(16))) _Float16 v16h;
typedef __attribute__((ext_vector_type(8)))  float    v8f;

#define BATCH 2048
#define HH 7
#define WW 24
#define NPOS 168          // 7*24
#define PADW 26
#define PADP 234          // 9*26

union FragU { int4 q[2]; v16h v; };

__device__ __forceinline__ float sigm_(float x) { return 1.0f / (1.0f + __expf(-x)); }
__device__ __forceinline__ float tanh_(float x) { return tanhf(x); }

// ---------------------------------------------------------------------------
// Implicit-GEMM conv accumulation, register-blocked over TWO M-tiles so each
// B (weight) fragment fetched from L2 feeds two WMMAs per gate.
// A from LDS (padded image, [pos][ch] f16); wf: [tap][kc][ntile][lane][16h]
// ---------------------------------------------------------------------------
template<int C, int CH>
__device__ __forceinline__ void conv_accum2(const _Float16* __restrict__ s,
                                            const _Float16* __restrict__ wf,
                                            int py0, int px0, int py1, int px1,
                                            int lhalf, int lane, int cotile,
                                            v8f acc0[4], v8f acc1[4])
{
    constexpr int KC = C / 32;
    constexpr int NT = (4 * CH) / 16;
#pragma unroll
    for (int tap = 0; tap < 9; ++tap) {
        const int dy = tap / 3, dx = tap % 3;
        const int pidx0 = (py0 + dy) * PADW + (px0 + dx);
        const int pidx1 = (py1 + dy) * PADW + (px1 + dx);
#pragma unroll
        for (int kc = 0; kc < KC; ++kc) {
            const int koff = kc * 32 + lhalf * 8;
            const int4* ap0 = (const int4*)(s + pidx0 * C + koff);
            const int4* ap1 = (const int4*)(s + pidx1 * C + koff);
            FragU A0, A1;
            A0.q[0] = ap0[0];   // halves [0,8)   -> K = lhalf*8 + 0..7
            A0.q[1] = ap0[2];   // halves [16,24) -> K = 16 + lhalf*8 + 0..7
            A1.q[0] = ap1[0];
            A1.q[1] = ap1[2];
#pragma unroll
            for (int g = 0; g < 4; ++g) {
                const int nt = g * (CH / 16) + cotile;
                const int4* bp = (const int4*)(wf +
                    ((size_t)(((tap * KC + kc) * NT + nt) * 32 + lane)) * 16);
                FragU B;
                B.q[0] = bp[0];
                B.q[1] = bp[1];
                acc0[g] = __builtin_amdgcn_wmma_f32_16x16x32_f16(
                    false, A0.v, false, B.v, (short)0, acc0[g], false, false);
                acc1[g] = __builtin_amdgcn_wmma_f32_16x16x32_f16(
                    false, A1.v, false, B.v, (short)0, acc1[g], false, false);
            }
        }
    }
}

// ---------------------------------------------------------------------------
// One ConvLSTM cell step for one batch image per workgroup.
// xin: [B][168][CIN] f16, h: [B][168][CH] f16 (in/out), c: [B][168][CH] f32
// wc: [3][CH][168] f32 peepholes, bx: [4*CH] f32
// ---------------------------------------------------------------------------
template<int CIN, int CH>
__global__ __launch_bounds__(256)
void convlstm_cell(const _Float16* __restrict__ xin,
                   _Float16* __restrict__ h,
                   float* __restrict__ c,
                   const _Float16* __restrict__ wxf,
                   const _Float16* __restrict__ whf,
                   const float* __restrict__ bx,
                   const float* __restrict__ wc)
{
    constexpr int COT = CH / 16;
    __shared__ __align__(16) _Float16 sx[PADP * CIN];
    __shared__ __align__(16) _Float16 sh[PADP * CH];

    const int b   = blockIdx.x;
    const int tid = threadIdx.x;
    const _Float16* xb = xin + (size_t)b * NPOS * CIN;
    _Float16*       hb = h   + (size_t)b * NPOS * CH;
    float*          cb = c   + (size_t)b * NPOS * CH;

    // ---- Stage halo-padded images into LDS, vectorized (b128 granules) ----
    {
        int4* sx4 = (int4*)sx;
        int4* sh4 = (int4*)sh;
        const int4 z4 = make_int4(0, 0, 0, 0);
        for (int i = tid; i < PADP * CIN / 8; i += 256) sx4[i] = z4;
        for (int i = tid; i < PADP * CH  / 8; i += 256) sh4[i] = z4;
        __syncthreads();
        // interior rows are contiguous in both layouts: 24*C halves per row
        const int4* gx = (const int4*)xb;
        const int4* gh = (const int4*)hb;
        for (int i = tid; i < 7 * 3 * CIN; i += 256) {       // 3*C int4 per row
            const int y = i / (3 * CIN);
            const int j = i - y * (3 * CIN);
            sx4[(((y + 1) * PADW + 1) * CIN) / 8 + j] = gx[y * 3 * CIN + j];
        }
        for (int i = tid; i < 7 * 3 * CH; i += 256) {
            const int y = i / (3 * CH);
            const int j = i - y * (3 * CH);
            sh4[(((y + 1) * PADW + 1) * CH) / 8 + j] = gh[y * 3 * CH + j];
        }
    }
    __syncthreads();

    const int wave  = tid >> 5;
    const int lane  = tid & 31;
    const int lhalf = lane >> 4;
    const int lmod  = lane & 15;
    const int NITEMS = 6 * COT;   // 6 M-tile pairs (12 tiles cover 168+pad rows)

    for (int item = wave; item < NITEMS; item += 8) {
        const int pair   = item / COT;
        const int cotile = item % COT;
        const int mtile0 = pair * 2;
        const int mtile1 = pair * 2 + 1;
        const int m0 = mtile0 * 16 + lmod;
        const int m1 = mtile1 * 16 + lmod;
        const int p0 = (m0 < NPOS) ? m0 : 0;   // clamp; masked at store
        const int p1 = (m1 < NPOS) ? m1 : 0;
        const int py0 = p0 / WW, px0 = p0 % WW;
        const int py1 = p1 / WW, px1 = p1 % WW;

        v8f acc0[4] = {};
        v8f acc1[4] = {};
        conv_accum2<CIN, CH>(sx, wxf, py0, px0, py1, px1, lhalf, lane, cotile, acc0, acc1);
        conv_accum2<CH,  CH>(sh, whf, py0, px0, py1, px1, lhalf, lane, cotile, acc0, acc1);

        // ---- LSTM epilogue (bias + peephole + gates), pure VALU ----
        const int co = cotile * 16 + lmod;
        const float b_i = bx[0 * CH + co];
        const float b_f = bx[1 * CH + co];
        const float b_c = bx[2 * CH + co];
        const float b_o = bx[3 * CH + co];
#pragma unroll
        for (int half = 0; half < 2; ++half) {
            const int mtile = half ? mtile1 : mtile0;
            const v8f* acc = half ? acc1 : acc0;
#pragma unroll
            for (int r = 0; r < 8; ++r) {
                const int mr = mtile * 16 + r + lhalf * 8;
                if (mr >= NPOS) continue;
                const size_t off = (size_t)mr * CH + co;
                const float cold = cb[off];
                const float wci = wc[(0 * CH + co) * NPOS + mr];
                const float wcf = wc[(1 * CH + co) * NPOS + mr];
                const float wco = wc[(2 * CH + co) * NPOS + mr];
                const float gi = acc[0][r] + b_i;
                const float gf = acc[1][r] + b_f;
                const float gc = acc[2][r] + b_c;
                const float go = acc[3][r] + b_o;
                const float ii = sigm_(gi + cold * wci);
                const float ff = sigm_(gf + cold * wcf);
                const float cc = ff * cold + ii * tanh_(gc);
                const float oo = sigm_(go + cc * wco);
                const float hh2 = oo * tanh_(cc);
                cb[off] = cc;
                hb[off] = (_Float16)hh2;
            }
        }
    }
}

// ---------------------------------------------------------------------------
// FC1: out[b, 0:64] = relu(h1_flat @ fc1_w.T + fc1_b), M=2048 K=10752 N=64
// 2-stage software pipeline: next fragments in flight during current WMMA.
// ---------------------------------------------------------------------------
__global__ __launch_bounds__(256)
void fc1_wmma(const _Float16* __restrict__ h1,
              const _Float16* __restrict__ wf,
              const float* __restrict__ bias,
              float* __restrict__ out)
{
    const int wave = threadIdx.x >> 5, lane = threadIdx.x & 31;
    const int item = blockIdx.x * 8 + wave;      // 512 items
    if (item >= 512) return;
    const int mtile = item >> 2, ntile = item & 3;
    const int lhalf = lane >> 4, lmod = lane & 15;
    const int m = mtile * 16 + lmod;
    const _Float16* arow = h1 + (size_t)m * 10752 + lhalf * 8;
    const _Float16* wlan = wf + (size_t)(ntile * 32 + lane) * 16;

    v8f acc = {};
    FragU A, B;
    {
        const int4* ap = (const int4*)arow;
        A.q[0] = ap[0]; A.q[1] = ap[2];
        const int4* bp = (const int4*)wlan;
        B.q[0] = bp[0]; B.q[1] = bp[1];
    }
    for (int kc = 0; kc < 336; ++kc) {
        FragU An = A, Bn = B;
        if (kc + 1 < 336) {
            const int4* ap = (const int4*)(arow + (kc + 1) * 32);
            An.q[0] = ap[0]; An.q[1] = ap[2];
            const int4* bp = (const int4*)(wlan + (size_t)(kc + 1) * 4 * 32 * 16);
            Bn.q[0] = bp[0]; Bn.q[1] = bp[1];
        }
        acc = __builtin_amdgcn_wmma_f32_16x16x32_f16(
            false, A.v, false, B.v, (short)0, acc, false, false);
        A = An; B = Bn;
    }
    const int n = ntile * 16 + lmod;
    const float bn = bias[n];
#pragma unroll
    for (int r = 0; r < 8; ++r) {
        const int mr = mtile * 16 + r + lhalf * 8;
        const float v = acc[r] + bn;
        out[(size_t)mr * 128 + n] = v > 0.0f ? v : 0.0f;
    }
}

// out[b, 64:128] = relu(x_ex @ exfc_w.T + exfc_b)   (tiny, VALU)
__global__ void exfc_kernel(const float* __restrict__ x_ex,
                            const float* __restrict__ w,
                            const float* __restrict__ bvec,
                            float* __restrict__ out)
{
    const int idx = blockIdx.x * 256 + threadIdx.x;   // B*64
    if (idx >= BATCH * 64) return;
    const int bb = idx >> 6, n = idx & 63;
    float s = bvec[n];
#pragma unroll
    for (int k = 0; k < 24; ++k) s += x_ex[bb * 24 + k] * w[n * 24 + k];
    out[(size_t)bb * 128 + 64 + n] = s > 0.0f ? s : 0.0f;
}

// ---------------------------------------------------------------------------
// Packing kernels (run once per launch)
// ---------------------------------------------------------------------------
// input [B][2][7][24] f32 -> xh [B][168][32] f16 (channels 2..31 zero)
__global__ void pack_x_kernel(const float* __restrict__ in, _Float16* __restrict__ xh)
{
    const int idx = blockIdx.x * 256 + threadIdx.x;   // B*168*32
    if (idx >= BATCH * NPOS * 32) return;
    const int ch = idx & 31;
    const int bp = idx >> 5;
    const int b = bp / NPOS, p = bp % NPOS;
    float v = 0.0f;
    if (ch < 2) v = in[(size_t)b * (2 * NPOS) + ch * NPOS + p];
    xh[idx] = (_Float16)v;
}

// Conv weight [Cout][Cin][3][3] f32 -> frag [9][Cpad/32][Cout/16][32][16] f16
// B fragment per lane: N = lane%16, halves e -> K = (lane/16)*8 + e (e<8), 16 + (lane/16)*8 + (e-8) (e>=8)
__global__ void pack_conv_w(const float* __restrict__ W, _Float16* __restrict__ frag,
                            int Cout, int Cin, int Cpad)
{
    const int KC = Cpad / 32, NTt = Cout / 16;
    const int total = 9 * KC * NTt * 512;
    const int idx = blockIdx.x * 256 + threadIdx.x;
    if (idx >= total) return;
    const int e = idx & 15;
    const int lane = (idx >> 4) & 31;
    int rest = idx >> 9;
    const int ntile = rest % NTt; rest /= NTt;
    const int kc = rest % KC;
    const int tap = rest / KC;
    const int K = (e < 8) ? ((lane >> 4) * 8 + e) : (16 + (lane >> 4) * 8 + (e - 8));
    const int cch = kc * 32 + K;
    const int n = ntile * 16 + (lane & 15);
    float val = 0.0f;
    if (cch < Cin) val = W[((size_t)n * Cin + cch) * 9 + tap];
    frag[idx] = (_Float16)val;
}

// fc1_w [64][10752] (col = ch*168 + p) -> frag [336][4][32][16] f16 with
// K ordered as our h1 flatten k = p*64 + ch
__global__ void pack_fc1_w(const float* __restrict__ W, _Float16* __restrict__ frag)
{
    const int idx = blockIdx.x * 256 + threadIdx.x;   // 336*4*512
    if (idx >= 336 * 4 * 512) return;
    const int e = idx & 15;
    const int lane = (idx >> 4) & 31;
    const int rest = idx >> 9;
    const int ntile = rest & 3;
    const int kc = rest >> 2;
    const int K = (e < 8) ? ((lane >> 4) * 8 + e) : (16 + (lane >> 4) * 8 + (e - 8));
    const int kg = kc * 32 + K;
    const int ch = kg & 63, p = kg >> 6;
    const int n = ntile * 16 + (lane & 15);
    frag[idx] = (_Float16)W[(size_t)n * 10752 + ch * NPOS + p];
}

__global__ void zero_f32(float* __restrict__ p, size_t n)
{
    const size_t i = (size_t)blockIdx.x * 256 + threadIdx.x;
    if (i < n) p[i] = 0.0f;
}

// ---------------------------------------------------------------------------
extern "C" void kernel_launch(void* const* d_in, const int* in_sizes, int n_in,
                              void* d_out, int out_size, void* d_ws, size_t ws_size,
                              hipStream_t stream)
{
    const float* input  = (const float*)d_in[0];
    const float* x_ex   = (const float*)d_in[1];
    const float* Wx0    = (const float*)d_in[2];
    const float* bx0    = (const float*)d_in[3];
    const float* Wh0    = (const float*)d_in[4];
    const float* Wc0    = (const float*)d_in[5];
    const float* Wx1    = (const float*)d_in[6];
    const float* bx1    = (const float*)d_in[7];
    const float* Wh1    = (const float*)d_in[8];
    const float* Wc1    = (const float*)d_in[9];
    const float* fc1_w  = (const float*)d_in[10];
    const float* fc1_b  = (const float*)d_in[11];
    const float* exfc_w = (const float*)d_in[12];
    const float* exfc_b = (const float*)d_in[13];
    float* out = (float*)d_out;

    char* ws = (char*)d_ws;
    size_t off = 0;
    auto carve = [&](size_t bytes) -> char* {
        char* p = ws + off;
        off += (bytes + 255) & ~(size_t)255;
        return p;
    };

    // State buffers (contiguous so one zero-fill covers them all)
    const size_t h0_b = (size_t)BATCH * NPOS * 32 * 2;   // f16
    const size_t c0_b = (size_t)BATCH * NPOS * 32 * 4;   // f32
    const size_t h1_b = (size_t)BATCH * NPOS * 64 * 2;
    const size_t c1_b = (size_t)BATCH * NPOS * 64 * 4;
    _Float16* h0 = (_Float16*)carve(h0_b);
    float*    c0 = (float*)   carve(c0_b);
    _Float16* h1 = (_Float16*)carve(h1_b);
    float*    c1 = (float*)   carve(c1_b);
    _Float16* xh0  = (_Float16*)carve((size_t)BATCH * NPOS * 32 * 2);
    _Float16* wxf0 = (_Float16*)carve((size_t)9 * 1 * 8  * 512 * 2);
    _Float16* whf0 = (_Float16*)carve((size_t)9 * 1 * 8  * 512 * 2);
    _Float16* wxf1 = (_Float16*)carve((size_t)9 * 1 * 16 * 512 * 2);
    _Float16* whf1 = (_Float16*)carve((size_t)9 * 2 * 16 * 512 * 2);
    _Float16* fc1f = (_Float16*)carve((size_t)336 * 4 * 512 * 2);

    // 1. zero states (h0..c1 contiguous)
    {
        const size_t nflt = (h0_b + c0_b + h1_b + c1_b) / 4;
        const int blocks = (int)((nflt + 255) / 256);
        zero_f32<<<blocks, 256, 0, stream>>>((float*)h0, nflt);
    }
    // 2. pack input and weights
    pack_x_kernel<<<(BATCH * NPOS * 32 + 255) / 256, 256, 0, stream>>>(input, xh0);
    pack_conv_w<<<(9 * 1 * 8  * 512 + 255) / 256, 256, 0, stream>>>(Wx0, wxf0, 128, 2,  32);
    pack_conv_w<<<(9 * 1 * 8  * 512 + 255) / 256, 256, 0, stream>>>(Wh0, whf0, 128, 32, 32);
    pack_conv_w<<<(9 * 1 * 16 * 512 + 255) / 256, 256, 0, stream>>>(Wx1, wxf1, 256, 32, 32);
    pack_conv_w<<<(9 * 2 * 16 * 512 + 255) / 256, 256, 0, stream>>>(Wh1, whf1, 256, 64, 64);
    pack_fc1_w<<<(336 * 4 * 512 + 255) / 256, 256, 0, stream>>>(fc1_w, fc1f);

    // 3. 7 timesteps, 2 cells each (sequential dependence)
    for (int step = 0; step < 7; ++step) {
        convlstm_cell<32, 32><<<BATCH, 256, 0, stream>>>(xh0, h0, c0, wxf0, whf0, bx0, Wc0);
        convlstm_cell<32, 64><<<BATCH, 256, 0, stream>>>(h0,  h1, c1, wxf1, whf1, bx1, Wc1);
    }

    // 4. heads
    fc1_wmma<<<64, 256, 0, stream>>>(h1, fc1f, fc1_b, out);
    exfc_kernel<<<(BATCH * 64 + 255) / 256, 256, 0, stream>>>(x_ex, exfc_w, exfc_b, out);
}
